// GCN_9972914061648
// MI455X (gfx1250) — matrix operand order repaired and checked by reading, not compiled, not run on hardware
//
#include <hip/hip_runtime.h>

typedef __attribute__((ext_vector_type(16))) _Float16 v16h;
typedef __attribute__((ext_vector_type(8)))  _Float16 v8h;
typedef __attribute__((ext_vector_type(8)))  float    v8f;

#define IN_F  256
#define H_F   64
#define OUT_F 16

// ---------------------------------------------------------------------------
// Degree / normalization
// ---------------------------------------------------------------------------
__global__ void init_deg_kernel(float* deg, int n) {
    int i = blockIdx.x * blockDim.x + threadIdx.x;
    if (i < n) deg[i] = 1.0f;               // self loop contributes 1
}

__global__ void accum_deg_kernel(const int* __restrict__ dst, float* deg, int e) {
    int i = blockIdx.x * blockDim.x + threadIdx.x;
    if (i < e) atomicAdd(&deg[dst[i]], 1.0f);
}

__global__ void dinv_kernel(float* deg, int n) {
    int i = blockIdx.x * blockDim.x + threadIdx.x;
    if (i < n) deg[i] = rsqrtf(deg[i]);     // deg >= 1 always (self loops)
}

// ---------------------------------------------------------------------------
// Helpers: build WMMA fragments
// ---------------------------------------------------------------------------
__device__ __forceinline__ v16h load_b_frag(const _Float16* bp) {
    // 16 contiguous halves -> two ds_load_b128
    v8h lo = *(const v8h*)(bp);
    v8h hi = *(const v8h*)(bp + 8);
    return __builtin_shufflevector(lo, hi, 0, 1, 2, 3, 4, 5, 6, 7,
                                           8, 9, 10, 11, 12, 13, 14, 15);
}

__device__ __forceinline__ v16h load_a_frag(const float* ap) {
    // a[0..15] = ap[0..15]  (f32 -> f16)
    float4 f0 = *(const float4*)(ap);
    float4 f1 = *(const float4*)(ap + 4);
    float4 f2 = *(const float4*)(ap + 8);
    float4 f3 = *(const float4*)(ap + 12);
    v16h a;
    a[0]  = (_Float16)f0.x; a[1]  = (_Float16)f0.y;
    a[2]  = (_Float16)f0.z; a[3]  = (_Float16)f0.w;
    a[4]  = (_Float16)f1.x; a[5]  = (_Float16)f1.y;
    a[6]  = (_Float16)f1.z; a[7]  = (_Float16)f1.w;
    a[8]  = (_Float16)f2.x; a[9]  = (_Float16)f2.y;
    a[10] = (_Float16)f2.z; a[11] = (_Float16)f2.w;
    a[12] = (_Float16)f3.x; a[13] = (_Float16)f3.y;
    a[14] = (_Float16)f3.z; a[15] = (_Float16)f3.w;
    return a;
}

// ---------------------------------------------------------------------------
// GEMM1: h1[N,64] = x[N,256] @ W1[256,64], f16 WMMA / f32 accumulate.
// W1 staged TRANSPOSED in LDS; each wave owns one fixed 16-col tile and
// hoists all 8 B fragments into registers, then streams M tiles.
// A fragments are double-buffered so the next step's converts never write
// registers a pending WMMA is reading (kills the 4-NOP WAR hazard).
// ---------------------------------------------------------------------------
__global__ void gemm1_wmma_kernel(const float* __restrict__ x,
                                  const float* __restrict__ W1,
                                  float* __restrict__ h1, int n) {
    constexpr int WS = IN_F + 8;            // 264 halves = 528 B (16B aligned)
    __shared__ _Float16 wt[H_F * WS];       // ~33 KB
    for (int i = threadIdx.x; i < IN_F * H_F; i += blockDim.x) {
        const int k = i >> 6, col = i & 63; // W1 is [256][64] row-major
        wt[col * WS + k] = (_Float16)W1[i];
    }
    __syncthreads();

    const int lane  = threadIdx.x & 31;
    const int wave  = threadIdx.x >> 5;
    const int wpb   = blockDim.x >> 5;
    const int gwave = blockIdx.x * wpb + wave;
    const int nwave = gridDim.x * wpb;

    const int nMTiles = n >> 4;             // 100000 = 6250*16, no tail

    const int r    = lane & 15;
    const int kbA  = (lane < 16) ? 0 : 16;  // A: K-half select
    const int kbB  = (lane < 16) ? 0 : 16;  // B: K-half select
    const int roff = (lane < 16) ? 0 : 8;   // C/D: row-half select

    const int nt  = gwave & 3;              // fixed column tile per wave
    const int col = (nt << 4) + r;

    // Hoist B fragments for all 8 K-steps (loop-invariant over M)
    v16h bfr[8];
    #pragma unroll
    for (int ks = 0; ks < 8; ++ks)
        bfr[ks] = load_b_frag(&wt[col * WS + ks * 32 + kbB]);

    for (int mt = gwave >> 2; mt < nMTiles; mt += (nwave >> 2)) {
        const int M0 = mt << 4;
        const float* xrow = x + (size_t)(M0 + r) * IN_F;

        v8f c = {};
        // Software-pipelined A: load ks+1 before issuing WMMA for ks.
        v16h a0 = load_a_frag(xrow + kbA);
        v16h a1;
        #pragma unroll
        for (int ks = 0; ks < 8; ks += 2) {
            a1 = load_a_frag(xrow + (ks + 1) * 32 + kbA);
            c = __builtin_amdgcn_wmma_f32_16x16x32_f16(
                    false, a0, false, bfr[ks], (short)0, c, false, false);
            if (ks < 6)
                a0 = load_a_frag(xrow + (ks + 2) * 32 + kbA);
            c = __builtin_amdgcn_wmma_f32_16x16x32_f16(
                    false, a1, false, bfr[ks + 1], (short)0, c, false, false);
        }
        #pragma unroll
        for (int i = 0; i < 8; ++i)
            h1[(size_t)(M0 + roff + i) * H_F + col] = c[i];
    }
}

// ---------------------------------------------------------------------------
// Edge scatter layer 1: agg1[dst] += dinv[src] * h1[src]   (64 ch/edge)
// ---------------------------------------------------------------------------
__global__ void scatter1_kernel(const int* __restrict__ src,
                                const int* __restrict__ dst,
                                const float* __restrict__ h1,
                                const float* __restrict__ dinv,
                                float* agg, int e) {
    const int eid = blockIdx.x * 4 + (threadIdx.x >> 6);
    const int c   = threadIdx.x & 63;
    if (eid < e) {
        const int u = src[eid];
        const int v = dst[eid];
        const float wsrc = dinv[u];
        atomicAdd(&agg[(size_t)v * H_F + c], h1[(size_t)u * H_F + c] * wsrc);
    }
}

// ---------------------------------------------------------------------------
// Finalize layer1 (in-place, agg1 -> z1):
//   z1 = relu(dinv[v]*agg + dinv[v]^2*h1[v] + b1)
// ---------------------------------------------------------------------------
__global__ void finalize1_kernel(float* agg1, const float* __restrict__ h1,
                                 const float* __restrict__ dinv,
                                 const float* __restrict__ b1, int n) {
    const int idx = blockIdx.x * blockDim.x + threadIdx.x;
    if (idx < n * H_F) {
        const int v = idx >> 6;
        const int c = idx & 63;
        const float di = dinv[v];
        const float z = di * agg1[idx] + di * di * h1[idx] + b1[c];
        agg1[idx] = fmaxf(z, 0.0f);
    }
}

// ---------------------------------------------------------------------------
// GEMM2: h2[N,16] = z1[N,64] @ W2[64,16], K=64 -> 2 WMMAs per tile.
// Both A fragments preloaded before the WMMAs (no WAR hazard).
// ---------------------------------------------------------------------------
__global__ void gemm2_wmma_kernel(const float* __restrict__ z1,
                                  const float* __restrict__ W2,
                                  float* __restrict__ h2, int n) {
    constexpr int WS = H_F + 8;             // 72 halves = 144 B (16B aligned)
    __shared__ _Float16 wt[OUT_F * WS];
    for (int i = threadIdx.x; i < H_F * OUT_F; i += blockDim.x) {
        const int k = i >> 4, col = i & 15; // W2 is [64][16] row-major
        wt[col * WS + k] = (_Float16)W2[i];
    }
    __syncthreads();

    const int lane  = threadIdx.x & 31;
    const int wave  = threadIdx.x >> 5;
    const int wpb   = blockDim.x >> 5;
    const int gwave = blockIdx.x * wpb + wave;
    const int nwave = gridDim.x * wpb;

    const int nMTiles = n >> 4;

    const int r    = lane & 15;
    const int kbA  = (lane < 16) ? 0 : 16;
    const int kbB  = (lane < 16) ? 0 : 16;
    const int roff = (lane < 16) ? 0 : 8;

    v16h bfr0 = load_b_frag(&wt[r * WS + kbB]);
    v16h bfr1 = load_b_frag(&wt[r * WS + 32 + kbB]);

    for (int mt = gwave; mt < nMTiles; mt += nwave) {
        const int M0 = mt << 4;
        const float* zrow = z1 + (size_t)(M0 + r) * H_F;

        v16h a0 = load_a_frag(zrow + kbA);
        v16h a1 = load_a_frag(zrow + 32 + kbA);

        v8f c = {};
        c = __builtin_amdgcn_wmma_f32_16x16x32_f16(
                false, a0, false, bfr0, (short)0, c, false, false);
        c = __builtin_amdgcn_wmma_f32_16x16x32_f16(
                false, a1, false, bfr1, (short)0, c, false, false);

        #pragma unroll
        for (int i = 0; i < 8; ++i)
            h2[(size_t)(M0 + roff + i) * OUT_F + r] = c[i];
    }
}

// ---------------------------------------------------------------------------
// Edge scatter layer 2: out[dst] += dinv[src] * h2[src]  (16 ch/edge)
// ---------------------------------------------------------------------------
__global__ void scatter2_kernel(const int* __restrict__ src,
                                const int* __restrict__ dst,
                                const float* __restrict__ h2,
                                const float* __restrict__ dinv,
                                float* out, int e) {
    const int eid = blockIdx.x * 16 + (threadIdx.x >> 4);
    const int c   = threadIdx.x & 15;
    if (eid < e) {
        const int u = src[eid];
        const int v = dst[eid];
        const float wsrc = dinv[u];
        atomicAdd(&out[(size_t)v * OUT_F + c], h2[(size_t)u * OUT_F + c] * wsrc);
    }
}

// ---------------------------------------------------------------------------
// Finalize layer2 (in-place on out): out = dinv*out + dinv^2*h2 + b2
// ---------------------------------------------------------------------------
__global__ void finalize2_kernel(float* out, const float* __restrict__ h2,
                                 const float* __restrict__ dinv,
                                 const float* __restrict__ b2, int n) {
    const int idx = blockIdx.x * blockDim.x + threadIdx.x;
    if (idx < n * OUT_F) {
        const int v = idx >> 4;
        const int c = idx & 15;
        const float di = dinv[v];
        out[idx] = di * out[idx] + di * di * h2[idx] + b2[c];
    }
}

// ---------------------------------------------------------------------------
extern "C" void kernel_launch(void* const* d_in, const int* in_sizes, int n_in,
                              void* d_out, int out_size, void* d_ws, size_t ws_size,
                              hipStream_t stream) {
    const float* x   = (const float*)d_in[0];
    const int*   ei  = (const int*)  d_in[1];
    const float* W1  = (const float*)d_in[2];
    const float* b1  = (const float*)d_in[3];
    const float* W2  = (const float*)d_in[4];
    const float* b2  = (const float*)d_in[5];

    const int N = in_sizes[0] / IN_F;        // 100000
    const int E = in_sizes[1] / 2;           // 1000000
    const int* src = ei;
    const int* dst = ei + E;

    // Workspace layout (floats): dinv | h1 | agg1(->z1) | h2
    float* dinv = (float*)d_ws;
    float* h1   = dinv + (size_t)N;
    float* agg1 = h1   + (size_t)N * H_F;
    float* h2   = agg1 + (size_t)N * H_F;
    float* out  = (float*)d_out;

    // 1) degrees -> dinv
    init_deg_kernel<<<(N + 255) / 256, 256, 0, stream>>>(dinv, N);
    accum_deg_kernel<<<(E + 255) / 256, 256, 0, stream>>>(dst, dinv, E);
    dinv_kernel<<<(N + 255) / 256, 256, 0, stream>>>(dinv, N);

    // 2) zero accumulators (graph-capture-safe async memsets)
    hipMemsetAsync(agg1, 0, (size_t)N * H_F * sizeof(float), stream);
    hipMemsetAsync(out,  0, (size_t)N * OUT_F * sizeof(float), stream);

    // 3) layer 1 transform (WMMA)
    gemm1_wmma_kernel<<<256, 256, 0, stream>>>(x, W1, h1, N);

    // 4) layer 1 edge aggregation
    scatter1_kernel<<<(E + 3) / 4, 256, 0, stream>>>(src, dst, h1, dinv, agg1, E);

    // 5) fuse dst-scale + self loop + bias + relu  (agg1 becomes z1)
    finalize1_kernel<<<((N * H_F) + 255) / 256, 256, 0, stream>>>(agg1, h1, dinv, b1, N);

    // 6) layer 2 transform (WMMA)
    gemm2_wmma_kernel<<<128, 256, 0, stream>>>(agg1, W2, h2, N);

    // 7) layer 2 edge aggregation directly into output
    scatter2_kernel<<<(E + 15) / 16, 256, 0, stream>>>(src, dst, h2, dinv, out, E);

    // 8) finalize output
    finalize2_kernel<<<((N * OUT_F) + 255) / 256, 256, 0, stream>>>(out, h2, dinv, b2, N);
}
